// MMoE_19954418057429
// MI455X (gfx1250) — matrix operand order repaired and checked by reading, not compiled, not run on hardware
//
#include <hip/hip_runtime.h>
#include <hip/hip_bf16.h>
#include <math.h>

typedef __attribute__((ext_vector_type(16))) __bf16       v16bf;
typedef __attribute__((ext_vector_type(8)))  float        v8f;
typedef __attribute__((ext_vector_type(4)))  unsigned int u32x4;

union Frag { v16bf v; u32x4 u[2]; };

#define NB    16384
#define DIN   1024
#define DHID  1024
#define DEXP  512
#define NE    8
#define NTASK 3
#define GIN   1088
#define GH    32
#define MTILE 32
#define HSTR  1032   // bf16 row stride: 1032/2=516 dwords, 516%64=4 -> conflict-free b128 reads

__device__ __forceinline__ float gelu_erf(float x) {
  return 0.5f * x * (1.0f + erff(x * 0.7071067811865476f));
}

// ---------------- prep: concat + fp32->bf16 ----------------
__global__ __launch_bounds__(256)
void cvt_combined(const float* __restrict__ hv, const float* __restrict__ ha,
                  __bf16* __restrict__ comb) {
  size_t i = (size_t)blockIdx.x * 256 + threadIdx.x;     // over NB*DIN
  int b = (int)(i >> 10), c = (int)(i & 1023);
  float x = (c < 512) ? hv[(size_t)b * 512 + c] : ha[(size_t)b * 512 + (c - 512)];
  comb[i] = (__bf16)x;
}

// ---------------- prep: transpose + fp32->bf16 (src[k][n] -> dst[n][k]) ----------------
__global__ __launch_bounds__(256)
void transpose_cvt(const float* __restrict__ src, __bf16* __restrict__ dst, int K, int N) {
  __shared__ float t[32][33];
  int n0 = blockIdx.x * 32, k0 = blockIdx.y * 32;
  size_t base = (size_t)blockIdx.z * K * N;
  for (int j = threadIdx.y; j < 32; j += 8)
    t[j][threadIdx.x] = src[base + (size_t)(k0 + j) * N + n0 + threadIdx.x];
  __syncthreads();
  for (int j = threadIdx.y; j < 32; j += 8)
    dst[base + (size_t)(n0 + j) * K + k0 + threadIdx.x] = (__bf16)t[threadIdx.x][j];
}

// ---------------- gates: fp32 GEMV -> GELU -> GEMV -> softmax ----------------
__global__ __launch_bounds__(128)
void gate_kernel(const float* __restrict__ hv, const float* __restrict__ ha,
                 const int* __restrict__ cid, const float* __restrict__ emb,
                 const float* __restrict__ Gw1, const float* __restrict__ Gb1,
                 const float* __restrict__ Gw2, const float* __restrict__ Gb2,
                 float* __restrict__ gatew) {
  __shared__ float gin[GIN];
  __shared__ float gh[NTASK * GH];
  __shared__ float lg[NTASK * NE];
  int b = blockIdx.x, tid = threadIdx.x;
  for (int i = tid; i < 512; i += 128) gin[i]       = hv[(size_t)b * 512 + i];
  for (int i = tid; i < 512; i += 128) gin[512 + i] = ha[(size_t)b * 512 + i];
  if (tid < 64) gin[1024 + tid] = emb[(size_t)cid[b] * 64 + tid];
  __syncthreads();
  if (tid < NTASK * GH) {
    int t = tid >> 5, j = tid & 31;
    float acc = Gb1[t * GH + j];
    const float* w = Gw1 + (size_t)t * GIN * GH + j;
    for (int i = 0; i < GIN; ++i) acc += gin[i] * w[(size_t)i * GH];
    gh[tid] = gelu_erf(acc);
  }
  __syncthreads();
  if (tid < NTASK * NE) {
    int t = tid >> 3, e2 = tid & 7;
    float acc = Gb2[t * NE + e2];
    const float* w = Gw2 + (size_t)t * GH * NE + e2;
    for (int j = 0; j < GH; ++j) acc += gh[t * GH + j] * w[j * NE];
    lg[tid] = acc;
  }
  __syncthreads();
  if (tid < NTASK) {
    float mx = -1e30f;
    for (int e2 = 0; e2 < NE; ++e2) mx = fmaxf(mx, lg[tid * NE + e2]);
    float ex[NE]; float s = 0.f;
    for (int e2 = 0; e2 < NE; ++e2) { ex[e2] = expf(lg[tid * NE + e2] - mx); s += ex[e2]; }
    float inv = 1.f / s;
    for (int e2 = 0; e2 < NE; ++e2)
      gatew[(size_t)b * (NTASK * NE) + tid * NE + e2] = ex[e2] * inv;
  }
}

// ---------------- fused experts + mixture (WMMA bf16, M-tile 32, 16 waves) ----------------
__global__ __launch_bounds__(512)
void moe_expert_kernel(const __bf16* __restrict__ comb,
                       const __bf16* __restrict__ W1T,
                       const __bf16* __restrict__ W2T,
                       const float* __restrict__ b1,
                       const float* __restrict__ lng,
                       const float* __restrict__ lnb,
                       const float* __restrict__ b2,
                       const float* __restrict__ gatew,
                       float* __restrict__ out) {
  __shared__ __bf16 h_lds[MTILE * HSTR];   // 66,048 B hidden tile (bf16, padded rows)
  __shared__ float red[2][MTILE];          // LN sum / sumsq -> mean / rstd
  __shared__ float wl[NTASK][MTILE];       // gate weights for this (tile, expert)

  const int tid  = threadIdx.x;
  const int wave = tid >> 5;               // 0..15
  const int lm   = tid & 15;               // lane & 15
  const int hi   = (tid >> 4) & 1;         // lane >> 4
  const int row0 = blockIdx.x * MTILE;

  v8f outacc[NTASK][2][2];                 // [task][mb][nt2]
  #pragma unroll
  for (int t = 0; t < NTASK; ++t)
    #pragma unroll
    for (int mb = 0; mb < 2; ++mb)
      #pragma unroll
      for (int nt = 0; nt < 2; ++nt)
        #pragma unroll
        for (int r = 0; r < 8; ++r) outacc[t][mb][nt][r] = 0.f;

  const __bf16* arow0 = comb + (size_t)(row0 + lm) * DIN;
  const __bf16* arow1 = comb + (size_t)(row0 + 16 + lm) * DIN;

  for (int e = 0; e < NE; ++e) {
    __syncthreads();                       // prev iter done with wl / h_lds
    if (tid < NTASK * MTILE) {
      int m = tid & 31, t = tid >> 5;
      wl[t][m] = gatew[(size_t)(row0 + m) * (NTASK * NE) + t * NE + e];
    }
    if (tid < 2 * MTILE) red[tid >> 5][tid & 31] = 0.f;
    __syncthreads();

    // ---- GEMM1: (32 x 1024) = A(32x1024) * W1[e](1024x1024); B-frag reused 2x ----
    v8f acc1[2][4];
    #pragma unroll
    for (int mb = 0; mb < 2; ++mb)
      #pragma unroll
      for (int nt = 0; nt < 4; ++nt)
        #pragma unroll
        for (int r = 0; r < 8; ++r) acc1[mb][nt][r] = 0.f;

    const __bf16* w1e = W1T + (size_t)e * DHID * DIN;
    for (int kb = 0; kb < DIN; kb += 32) {
      Frag fa0, fa1;
      fa0.u[0] = *(const u32x4*)(arow0 + kb + hi * 8);
      fa0.u[1] = *(const u32x4*)(arow0 + kb + 16 + hi * 8);
      fa1.u[0] = *(const u32x4*)(arow1 + kb + hi * 8);
      fa1.u[1] = *(const u32x4*)(arow1 + kb + 16 + hi * 8);
      #pragma unroll
      for (int nt = 0; nt < 4; ++nt) {
        const __bf16* brow = w1e + (size_t)(wave * 64 + nt * 16 + lm) * DIN + kb + hi * 8;
        Frag fb;
        fb.u[0] = *(const u32x4*)(brow);
        fb.u[1] = *(const u32x4*)(brow + 16);
        acc1[0][nt] = __builtin_amdgcn_wmma_f32_16x16x32_bf16(
            false, fa0.v, false, fb.v, (short)0, acc1[0][nt], false, false);
        acc1[1][nt] = __builtin_amdgcn_wmma_f32_16x16x32_bf16(
            false, fa1.v, false, fb.v, (short)0, acc1[1][nt], false, false);
      }
    }

    // ---- bias + LayerNorm stats (row m = mb*16 + r + 8*hi) ----
    v8f s[2], q[2];
    #pragma unroll
    for (int mb = 0; mb < 2; ++mb)
      #pragma unroll
      for (int r = 0; r < 8; ++r) { s[mb][r] = 0.f; q[mb][r] = 0.f; }
    #pragma unroll
    for (int nt = 0; nt < 4; ++nt) {
      float bv = b1[e * DHID + wave * 64 + nt * 16 + lm];
      #pragma unroll
      for (int mb = 0; mb < 2; ++mb)
        #pragma unroll
        for (int r = 0; r < 8; ++r) {
          float x = acc1[mb][nt][r] + bv;
          acc1[mb][nt][r] = x; s[mb][r] += x; q[mb][r] += x * x;
        }
    }
    #pragma unroll
    for (int msk = 1; msk < 16; msk <<= 1)
      #pragma unroll
      for (int mb = 0; mb < 2; ++mb)
        #pragma unroll
        for (int r = 0; r < 8; ++r) {
          s[mb][r] += __shfl_xor(s[mb][r], msk, 32);
          q[mb][r] += __shfl_xor(q[mb][r], msk, 32);
        }
    if (lm == 0) {
      #pragma unroll
      for (int mb = 0; mb < 2; ++mb)
        #pragma unroll
        for (int r = 0; r < 8; ++r) {
          atomicAdd(&red[0][mb * 16 + r + 8 * hi], s[mb][r]);
          atomicAdd(&red[1][mb * 16 + r + 8 * hi], q[mb][r]);
        }
    }
    __syncthreads();
    if (tid < MTILE) {
      float mu  = red[0][tid] * (1.0f / DHID);
      float var = red[1][tid] * (1.0f / DHID) - mu * mu;
      red[0][tid] = mu;
      red[1][tid] = rsqrtf(var + 1e-5f);
    }
    __syncthreads();
    float mu[2][8], rs[2][8];
    #pragma unroll
    for (int mb = 0; mb < 2; ++mb)
      #pragma unroll
      for (int r = 0; r < 8; ++r) {
        mu[mb][r] = red[0][mb * 16 + r + 8 * hi];
        rs[mb][r] = red[1][mb * 16 + r + 8 * hi];
      }

    // ---- LN*g+b -> GELU -> bf16 hidden tile in LDS ----
    #pragma unroll
    for (int nt = 0; nt < 4; ++nt) {
      int n = wave * 64 + nt * 16 + lm;
      float g  = lng[e * DHID + n];
      float bb = lnb[e * DHID + n];
      #pragma unroll
      for (int mb = 0; mb < 2; ++mb)
        #pragma unroll
        for (int r = 0; r < 8; ++r) {
          float x = (acc1[mb][nt][r] - mu[mb][r]) * rs[mb][r] * g + bb;
          h_lds[(mb * 16 + r + 8 * hi) * HSTR + n] = (__bf16)gelu_erf(x);
        }
    }
    __syncthreads();

    // ---- GEMM2: (32 x 512) = H(32x1024) * W2[e](1024x512); B-frag reused 2x ----
    v8f acc2[2][2];
    #pragma unroll
    for (int mb = 0; mb < 2; ++mb)
      #pragma unroll
      for (int nt = 0; nt < 2; ++nt)
        #pragma unroll
        for (int r = 0; r < 8; ++r) acc2[mb][nt][r] = 0.f;

    const __bf16* w2e   = W2T + (size_t)e * DEXP * DHID;
    const __bf16* hrow0 = h_lds + lm * HSTR;
    const __bf16* hrow1 = h_lds + (16 + lm) * HSTR;
    for (int kb = 0; kb < DHID; kb += 32) {
      Frag ha0, ha1;
      ha0.u[0] = *(const u32x4*)(hrow0 + kb + hi * 8);
      ha0.u[1] = *(const u32x4*)(hrow0 + kb + 16 + hi * 8);
      ha1.u[0] = *(const u32x4*)(hrow1 + kb + hi * 8);
      ha1.u[1] = *(const u32x4*)(hrow1 + kb + 16 + hi * 8);
      #pragma unroll
      for (int nt = 0; nt < 2; ++nt) {
        const __bf16* brow = w2e + (size_t)(wave * 32 + nt * 16 + lm) * DHID + kb + hi * 8;
        Frag fb;
        fb.u[0] = *(const u32x4*)(brow);
        fb.u[1] = *(const u32x4*)(brow + 16);
        acc2[0][nt] = __builtin_amdgcn_wmma_f32_16x16x32_bf16(
            false, ha0.v, false, fb.v, (short)0, acc2[0][nt], false, false);
        acc2[1][nt] = __builtin_amdgcn_wmma_f32_16x16x32_bf16(
            false, ha1.v, false, fb.v, (short)0, acc2[1][nt], false, false);
      }
    }

    // ---- bias + GELU + weighted mixture accumulate ----
    float wv[NTASK][2][8];
    #pragma unroll
    for (int t = 0; t < NTASK; ++t)
      #pragma unroll
      for (int mb = 0; mb < 2; ++mb)
        #pragma unroll
        for (int r = 0; r < 8; ++r) wv[t][mb][r] = wl[t][mb * 16 + r + 8 * hi];
    #pragma unroll
    for (int nt = 0; nt < 2; ++nt) {
      float bv = b2[e * DEXP + wave * 32 + nt * 16 + lm];
      #pragma unroll
      for (int mb = 0; mb < 2; ++mb)
        #pragma unroll
        for (int r = 0; r < 8; ++r) {
          float x = gelu_erf(acc2[mb][nt][r] + bv);
          #pragma unroll
          for (int t = 0; t < NTASK; ++t) outacc[t][mb][nt][r] += wv[t][mb][r] * x;
        }
    }
  }

  // ---- store per-task outputs (concatenated in return order) ----
  #pragma unroll
  for (int t = 0; t < NTASK; ++t)
    #pragma unroll
    for (int mb = 0; mb < 2; ++mb)
      #pragma unroll
      for (int nt = 0; nt < 2; ++nt) {
        int n = wave * 32 + nt * 16 + lm;
        #pragma unroll
        for (int r = 0; r < 8; ++r)
          out[(size_t)t * NB * DEXP + (size_t)(row0 + mb * 16 + r + 8 * hi) * DEXP + n] =
              outacc[t][mb][nt][r];
      }
}

extern "C" void kernel_launch(void* const* d_in, const int* in_sizes, int n_in,
                              void* d_out, int out_size, void* d_ws, size_t ws_size,
                              hipStream_t stream) {
  const float* h_val = (const float*)d_in[0];
  const float* h_aro = (const float*)d_in[1];
  const int*   cid   = (const int*)  d_in[2];
  const float* W1    = (const float*)d_in[3];
  const float* b1    = (const float*)d_in[4];
  const float* ln_g  = (const float*)d_in[5];
  const float* ln_b  = (const float*)d_in[6];
  const float* W2    = (const float*)d_in[7];
  const float* b2    = (const float*)d_in[8];
  const float* emb   = (const float*)d_in[9];
  const float* Gw1   = (const float*)d_in[10];
  const float* Gb1   = (const float*)d_in[11];
  const float* Gw2   = (const float*)d_in[12];
  const float* Gb2   = (const float*)d_in[13];
  float* out = (float*)d_out;

  // workspace layout (bytes)
  char* ws = (char*)d_ws;
  __bf16* comb  = (__bf16*)(ws);                       // 16384*1024*2 = 33,554,432
  __bf16* W1T   = (__bf16*)(ws + 33554432);            // 8*1024*1024*2 = 16,777,216
  __bf16* W2T   = (__bf16*)(ws + 50331648);            // 8*512*1024*2  =  8,388,608
  float*  gatew = (float*) (ws + 58720256);            // 16384*3*8*4   =  1,572,864

  cvt_combined<<<(NB * DIN) / 256, 256, 0, stream>>>(h_val, h_aro, comb);
  dim3 tb(32, 8);
  transpose_cvt<<<dim3(DHID / 32, DIN / 32, NE), tb, 0, stream>>>(W1, W1T, DIN, DHID);
  transpose_cvt<<<dim3(DEXP / 32, DHID / 32, NE), tb, 0, stream>>>(W2, W2T, DHID, DEXP);
  gate_kernel<<<NB, 128, 0, stream>>>(h_val, h_aro, cid, emb, Gw1, Gb1, Gw2, Gb2, gatew);
  moe_expert_kernel<<<NB / MTILE, 512, 0, stream>>>(comb, W1T, W2T, b1, ln_g, ln_b, b2,
                                                    gatew, out);
}
